// Attention_1709396983939
// MI455X (gfx1250) — matrix operand order repaired and checked by reading, not compiled, not run on hardware
//
#include <hip/hip_runtime.h>
#include <hip/hip_bf16.h>

// Problem dims (fixed by the reference)
constexpr int NB  = 4;
constexpr int NL  = 2048;
constexpr int ND  = 1024;
constexpr int NH  = 16;
constexpr int NA  = 64;
constexpr int NHA = NH * NA;          // 1024
constexpr long NBL = (long)NB * NL;   // 8192

typedef __bf16 bf16;
typedef __attribute__((ext_vector_type(8)))  bf16      v8bf;
typedef __attribute__((ext_vector_type(16))) bf16      v16bf;
typedef __attribute__((ext_vector_type(8)))  float     v8f;
typedef __attribute__((ext_vector_type(4)))  unsigned  v4u;
typedef __attribute__((ext_vector_type(8)))  unsigned  v8u;

__device__ inline v16bf concat8(v8bf a, v8bf b) {
  return __builtin_shufflevector(a, b, 0,1,2,3,4,5,6,7,8,9,10,11,12,13,14,15);
}

__device__ inline v8f wmma_bf16(v16bf a, v16bf b, v8f c) {
  // D = A(16x32 bf16) * B(32x16 bf16) + C(16x16 f32)
  return __builtin_amdgcn_wmma_f32_16x16x32_bf16(
      /*neg_a=*/false, a, /*neg_b=*/false, b,
      /*c_mod=*/(short)0, c, /*reuse_a=*/false, /*reuse_b=*/false);
}

// ---------------------------------------------------------------------------
// Butterfly helpers over the 16-lane halves (wave32): single ds_swizzle with
// immediate XOR pattern (group-of-32 mode: and=0x1f, xor in bits [14:10]).
// ---------------------------------------------------------------------------
template <int XORM>
__device__ inline float swz_xor(float v) {
#if __has_builtin(__builtin_amdgcn_ds_swizzle)
  return __int_as_float(
      __builtin_amdgcn_ds_swizzle(__float_as_int(v), (XORM << 10) | 0x1f));
#else
  return __shfl_xor(v, XORM);
#endif
}
__device__ inline float redmax16(float v) {
  v = fmaxf(v, swz_xor<1>(v));
  v = fmaxf(v, swz_xor<2>(v));
  v = fmaxf(v, swz_xor<4>(v));
  v = fmaxf(v, swz_xor<8>(v));
  return v;
}
__device__ inline float redsum16(float v) {
  v += swz_xor<1>(v);
  v += swz_xor<2>(v);
  v += swz_xor<4>(v);
  v += swz_xor<8>(v);
  return v;
}

// ---------------------------------------------------------------------------
// Tensor Data Mover: 2D bf16 tile (tile1 rows x tile0 elems, row stride
// stride0 elems) from global -> LDS, contiguous at lds byte offset.
// D# per CDNA5 ISA. Issued once per wave (scalar);
// completion tracked with TENSORcnt (in-order per wave).
// ---------------------------------------------------------------------------
__device__ inline void tdm_load_2d_bf16(const void* g, unsigned lds_off,
                                        unsigned tile0, unsigned tile1,
                                        unsigned stride0) {
  unsigned long long ga = (unsigned long long)g;
  v4u g0;
  g0[0] = 1u;                                            // count = 1 (valid)
  g0[1] = lds_off;                                       // lds_addr (bytes)
  g0[2] = (unsigned)(ga & 0xffffffffull);                // global_addr[31:0]
  g0[3] = (unsigned)((ga >> 32) & 0x1ffffffull)          // global_addr[56:32]
          | 0x80000000u;                                 // type = 2 ("image")
  const unsigned td0 = stride0;                          // tensor_dim0 (elems)
  const unsigned td1 = 1u << 20;                         // tensor_dim1 (large)
  v8u g1;
  g1[0] = 1u << 16;                                      // data_size = 2 bytes
  g1[1] = (td0 & 0xffffu) << 16;                         // tensor_dim0[15:0]
  g1[2] = (td0 >> 16) | ((td1 & 0xffffu) << 16);         // td0[31:16] | td1[15:0]
  g1[3] = (td1 >> 16) | (tile0 << 16);                   // td1[31:16] | tile_dim0
  g1[4] = tile1;                                         // tile_dim1 (tile_dim2=0)
  g1[5] = stride0;                                       // tensor_dim0_stride[31:0]
  g1[6] = 0u;                                            // stride0[47:32] | stride1[15:0]
  g1[7] = 0u;                                            // stride1[47:16]
  asm volatile("tensor_load_to_lds %0, %1" :: "s"(g0), "s"(g1) : "memory");
}

__device__ inline unsigned lds_offset_of(const void* p) {
  // generic pointers to LDS carry the LDS byte address in the low 32 bits
  return (unsigned)(unsigned long long)(size_t)p;
}

// ---------------------------------------------------------------------------
// Utility kernels: fp32 -> bf16 casts / transposes
// ---------------------------------------------------------------------------
__global__ void cast_f32_bf16(const float* __restrict__ in, bf16* __restrict__ out, long n) {
  long i = (long)blockIdx.x * blockDim.x + threadIdx.x;
  long stride = (long)gridDim.x * blockDim.x;
  for (; i < n; i += stride) out[i] = (bf16)in[i];
}

// W[D][HA] (f32) -> Wt[HA][D] (bf16)
__global__ void cast_transpose_w(const float* __restrict__ in, bf16* __restrict__ out) {
  long n = (long)ND * NHA;
  long i = (long)blockIdx.x * blockDim.x + threadIdx.x;
  long stride = (long)gridDim.x * blockDim.x;
  for (; i < n; i += stride) {
    long d = i / NHA, c = i % NHA;
    out[c * (long)ND + d] = (bf16)in[i];
  }
}

// v[(b*L+l)*HA + h*A + a] -> vt[((b*H+h)*A + a)*L + l]   (all bf16)
__global__ void transpose_v(const bf16* __restrict__ v, bf16* __restrict__ vt) {
  long n = (long)NB * NL * NH * NA;
  long i = (long)blockIdx.x * blockDim.x + threadIdx.x;
  long stride = (long)gridDim.x * blockDim.x;
  for (; i < n; i += stride) {
    int a = (int)(i % NA);
    int h = (int)((i / NA) % NH);
    int l = (int)((i / (NA * NH)) % NL);
    int b = (int)(i / ((long)NA * NH * NL));
    vt[(((long)b * NH + h) * NA + a) * NL + l] = v[i];
  }
}

// ---------------------------------------------------------------------------
// Tiled bf16 GEMM:  C[M][N] = A[M][K] * Bt[N][K]^T   (Bt stored K-contiguous)
// 256 threads (8 waves). Tile 128x128x64; TDM staging two stages deep.
// Wave tile 64x32: per stage 2 K-steps x (4x2) WMMA 16x16x32 bf16, f32 accum.
// ---------------------------------------------------------------------------
template <bool OUT_BF16>
__global__ __launch_bounds__(256) void gemm_bf16(const bf16* __restrict__ Ag,
                                                 const bf16* __restrict__ Btg,
                                                 void* __restrict__ Cg,
                                                 int M, int N, int K) {
  __shared__ __align__(128) bf16 As[2][128 * 64];
  __shared__ __align__(128) bf16 Bs[2][128 * 64];

  const int t    = threadIdx.x;
  const int lane = t & 31;
  const int w    = t >> 5;
  const int wr   = w >> 2;      // 0..1  (64-row strip)
  const int wc   = w & 3;       // 0..3  (32-col strip)
  const int r    = lane & 15;
  const int hi   = lane >> 4;

  const long mbase = (long)blockIdx.y * 128;
  const long nbase = (long)blockIdx.x * 128;
  const bool issuer = (w == 0);

  const unsigned ldsA[2] = { lds_offset_of(&As[0][0]), lds_offset_of(&As[1][0]) };
  const unsigned ldsB[2] = { lds_offset_of(&Bs[0][0]), lds_offset_of(&Bs[1][0]) };

  v8f acc[4][2] = {};

  const int nk = K / 64;
  if (issuer) {   // two stages in flight
    tdm_load_2d_bf16(Ag  + mbase * (long)K,      ldsA[0], 64, 128, (unsigned)K);
    tdm_load_2d_bf16(Btg + nbase * (long)K,      ldsB[0], 64, 128, (unsigned)K);
    tdm_load_2d_bf16(Ag  + mbase * (long)K + 64, ldsA[1], 64, 128, (unsigned)K);
    tdm_load_2d_bf16(Btg + nbase * (long)K + 64, ldsB[1], 64, 128, (unsigned)K);
  }

  for (int i = 0; i < nk; ++i) {
    const int cur = i & 1;
    if (issuer)
      __builtin_amdgcn_s_wait_tensorcnt(2);  // in-order: stage i's 2 DMAs done
    __syncthreads();

#pragma unroll
    for (int s = 0; s < 2; ++s) {            // two WMMA K-steps per stage
      v16bf af[4], bfv[2];
#pragma unroll
      for (int tm = 0; tm < 4; ++tm) {
        const bf16* p = &As[cur][(wr * 64 + tm * 16 + r) * 64 + s * 32];
        v8bf c0 = *(const v8bf*)(p + 16 * hi);
        v8bf c1 = *(const v8bf*)(p + 16 * hi + 8);
        af[tm] = concat8(c0, c1);
      }
#pragma unroll
      for (int tn = 0; tn < 2; ++tn)
        bfv[tn] = *(const v16bf*)&Bs[cur][(wc * 32 + tn * 16 + r) * 64 + s * 32 + 16 * hi];

#pragma unroll
      for (int tm = 0; tm < 4; ++tm)
#pragma unroll
        for (int tn = 0; tn < 2; ++tn)
          acc[tm][tn] = wmma_bf16(af[tm], bfv[tn], acc[tm][tn]);
    }
    __syncthreads();

    if (issuer && i + 2 < nk) {              // refill freed buffer, 2 ahead
      const long kt = (long)(i + 2) * 64;
      tdm_load_2d_bf16(Ag  + mbase * (long)K + kt, ldsA[cur], 64, 128, (unsigned)K);
      tdm_load_2d_bf16(Btg + nbase * (long)K + kt, ldsB[cur], 64, 128, (unsigned)K);
    }
  }

#pragma unroll
  for (int tm = 0; tm < 4; ++tm)
#pragma unroll
    for (int tn = 0; tn < 2; ++tn)
#pragma unroll
      for (int j = 0; j < 8; ++j) {
        long row = mbase + wr * 64 + tm * 16 + hi * 8 + j;
        long col = nbase + wc * 32 + tn * 16 + r;
        float val = acc[tm][tn][j];
        if (OUT_BF16) ((bf16*)Cg)[row * N + col]  = (bf16)val;
        else          ((float*)Cg)[row * N + col] = val;
      }
}

// ---------------------------------------------------------------------------
// Flash attention: one (b, h, 128-query block) per workgroup.
// 8 waves x 16 query rows. Keys in steps of 32; K/V tiles staged via TDM,
// two stages deep, shared by all 8 waves.
// q,k : bf16 [B][L][H*A] ; vt : bf16 [B][H][A][L] ; ctx : bf16 [B][L][H*A]
// ---------------------------------------------------------------------------
__global__ __launch_bounds__(256) void flash_attn(const bf16* __restrict__ q,
                                                  const bf16* __restrict__ k,
                                                  const bf16* __restrict__ vt,
                                                  bf16* __restrict__ ctx) {
  __shared__ __align__(128) bf16 Ks[2][32 * 64];  // [key in tile][a]
  __shared__ __align__(128) bf16 Vs[2][64 * 32];  // [a][key in tile]
  __shared__ __align__(128) bf16 Ps[8 * 16 * 32]; // per-wave 16x32 P tile

  const int t    = threadIdx.x;
  const int lane = t & 31;
  const int w    = t >> 5;
  const int r    = lane & 15;
  const int hi   = lane >> 4;

  const int nLblk = NL / 128;
  const int bid  = blockIdx.x;
  const int lblk = bid % nLblk;
  const int h    = (bid / nLblk) % NH;
  const int b    = bid / (nLblk * NH);

  const long q0 = (long)lblk * 128 + w * 16;   // first query row of this wave
  const bool issuer = (w == 0);

  const bf16* kbase = k  + (long)b * NL * NHA + h * NA;         // [L][HA] slice
  const bf16* vbase = vt + ((long)b * NH + h) * (long)NA * NL;  // [A][L] slice

  const unsigned ldsK[2] = { lds_offset_of(&Ks[0][0]), lds_offset_of(&Ks[1][0]) };
  const unsigned ldsV[2] = { lds_offset_of(&Vs[0][0]), lds_offset_of(&Vs[1][0]) };

  // Q fragments for this wave's 16 rows (A = 64 -> two K-steps of 32)
  const bf16* qrow = q + ((long)b * NL + q0 + r) * NHA + h * NA;
  v16bf aq[2];
#pragma unroll
  for (int s = 0; s < 2; ++s) {
    v8bf c0 = *(const v8bf*)(qrow + s * 32 + 16 * hi);
    v8bf c1 = *(const v8bf*)(qrow + s * 32 + 16 * hi + 8);
    aq[s] = concat8(c0, c1);
  }

  float m[8], lsum[8];
#pragma unroll
  for (int j = 0; j < 8; ++j) { m[j] = -__builtin_huge_valf(); lsum[j] = 0.0f; }
  v8f o[4] = {};

  bf16* pw = &Ps[w * 16 * 32];

  if (issuer) {   // two 32-key stages in flight
    tdm_load_2d_bf16(kbase,            ldsK[0], 64, 32, NHA);
    tdm_load_2d_bf16(vbase,            ldsV[0], 32, 64, NL);
    tdm_load_2d_bf16(kbase + 32 * NHA, ldsK[1], 64, 32, NHA);
    tdm_load_2d_bf16(vbase + 32,       ldsV[1], 32, 64, NL);
  }

  const int nkb = NL / 32;
  for (int i = 0; i < nkb; ++i) {
    const int cur = i & 1;
    if (issuer)
      __builtin_amdgcn_s_wait_tensorcnt(2);  // stage i's K and V tiles are in
    __syncthreads();

    // ---- scores: S[16q x 32keys] = q . k^T (two 16-key tiles) ----
    v8f s0 = {}, s1 = {};
#pragma unroll
    for (int s = 0; s < 2; ++s) {
      v16bf bk0 = *(const v16bf*)&Ks[cur][(     r) * 64 + s * 32 + 16 * hi];
      v16bf bk1 = *(const v16bf*)&Ks[cur][(16 + r) * 64 + s * 32 + 16 * hi];
      s0 = wmma_bf16(aq[s], bk0, s0);
      s1 = wmma_bf16(aq[s], bk1, s1);
    }

    // ---- online softmax (rows live as element j, cols across 16 lanes) ----
    float mn[8], alpha[8], p0[8], p1[8];
#pragma unroll
    for (int j = 0; j < 8; ++j)
      mn[j] = fmaxf(m[j], redmax16(fmaxf(s0[j], s1[j])));
#pragma unroll
    for (int j = 0; j < 8; ++j) {
      p0[j] = __expf(s0[j] - mn[j]);
      p1[j] = __expf(s1[j] - mn[j]);
      float rs = redsum16(p0[j] + p1[j]);
      alpha[j] = __expf(m[j] - mn[j]);
      lsum[j] = lsum[j] * alpha[j] + rs;
      m[j] = mn[j];
    }
#pragma unroll
    for (int tn = 0; tn < 4; ++tn)
#pragma unroll
      for (int j = 0; j < 8; ++j) o[tn][j] *= alpha[j];

    // ---- P (16x32) to LDS in C-layout, reload as A-fragment ----
#pragma unroll
    for (int j = 0; j < 8; ++j) {
      int row = hi * 8 + j;
      pw[row * 32 + r]      = (bf16)p0[j];
      pw[row * 32 + 16 + r] = (bf16)p1[j];
    }
    // same-wave LDS RAW: DS ops are in-order per wave; compiler inserts waits
    v8bf c0 = *(const v8bf*)(pw + r * 32 + 16 * hi);
    v8bf c1 = *(const v8bf*)(pw + r * 32 + 16 * hi + 8);
    v16bf pa = concat8(c0, c1);

    // ---- ctx += P . V ----
#pragma unroll
    for (int tn = 0; tn < 4; ++tn) {
      v16bf bv = *(const v16bf*)&Vs[cur][(tn * 16 + r) * 32 + 16 * hi];
      o[tn] = wmma_bf16(pa, bv, o[tn]);
    }
    __syncthreads();

    if (issuer && i + 2 < nkb) {             // refill freed buffer, 2 ahead
      const long kb2 = (long)(i + 2) * 32;
      tdm_load_2d_bf16(kbase + kb2 * NHA, ldsK[cur], 64, 32, NHA);
      tdm_load_2d_bf16(vbase + kb2,       ldsV[cur], 32, 64, NL);
    }
  }

  // ---- normalize and write ctx (bf16, [B][L][H*A]) ----
#pragma unroll
  for (int tn = 0; tn < 4; ++tn)
#pragma unroll
    for (int j = 0; j < 8; ++j) {
      long row = (long)b * NL + q0 + hi * 8 + j;
      ctx[row * NHA + h * NA + tn * 16 + r] = (bf16)(o[tn][j] / lsum[j]);
    }
}

// ---------------------------------------------------------------------------
// Host-side launch
// ---------------------------------------------------------------------------
extern "C" void kernel_launch(void* const* d_in, const int* in_sizes, int n_in,
                              void* d_out, int out_size, void* d_ws, size_t ws_size,
                              hipStream_t stream) {
  const float* xq  = (const float*)d_in[0];
  const float* xkv = (const float*)d_in[1];
  const float* Qw  = (const float*)d_in[2];
  const float* Kw  = (const float*)d_in[3];
  const float* Vw  = (const float*)d_in[4];
  const float* Ow  = (const float*)d_in[5];
  float* out = (float*)d_out;

  bf16* ws = (bf16*)d_ws;
  const size_t E  = (size_t)NBL * ND;     // 8192*1024 activations
  const size_t W1 = (size_t)ND * NHA;     // 1024*1024 weights

  bf16* xq_bf  = ws;                // dead after q projection
  bf16* xkv_bf = ws + E;            // dead after k/v projections
  bf16* q_bf   = ws + 2 * E;
  bf16* k_bf   = ws + 3 * E;
  bf16* v_bf   = ws + 4 * E;
  bf16* Qt     = ws + 5 * E;
  bf16* Kt     = Qt + W1;
  bf16* Vt     = Qt + 2 * W1;
  bf16* Ob     = Qt + 3 * W1;       // O needs no transpose: Bt[n=d][k=ha] == O[d][ha]
  bf16* ctx_bf = xq_bf;             // reuse
  bf16* v_t    = xkv_bf;            // reuse

  cast_f32_bf16<<<2048, 256, 0, stream>>>(xq,  xq_bf,  (long)E);
  cast_f32_bf16<<<2048, 256, 0, stream>>>(xkv, xkv_bf, (long)E);
  cast_f32_bf16<<<512,  256, 0, stream>>>(Ow,  Ob,     (long)W1);
  cast_transpose_w<<<1024, 256, 0, stream>>>(Qw, Qt);
  cast_transpose_w<<<1024, 256, 0, stream>>>(Kw, Kt);
  cast_transpose_w<<<1024, 256, 0, stream>>>(Vw, Vt);

  dim3 gproj(NHA / 128, (unsigned)(NBL / 128));   // (8, 64)
  gemm_bf16<true><<<gproj, 256, 0, stream>>>(xq_bf,  Qt, q_bf, (int)NBL, NHA, ND);
  gemm_bf16<true><<<gproj, 256, 0, stream>>>(xkv_bf, Kt, k_bf, (int)NBL, NHA, ND);
  gemm_bf16<true><<<gproj, 256, 0, stream>>>(xkv_bf, Vt, v_bf, (int)NBL, NHA, ND);

  transpose_v<<<4096, 256, 0, stream>>>(v_bf, v_t);

  flash_attn<<<NB * NH * (NL / 128), 256, 0, stream>>>(q_bf, k_bf, v_t, ctx_bf);

  dim3 gout(ND / 128, (unsigned)(NBL / 128));
  gemm_bf16<false><<<gout, 256, 0, stream>>>(ctx_bf, Ob, out, (int)NBL, ND, NHA);
}
